// SwinTransformerLayer_87428354277384
// MI455X (gfx1250) — matrix operand order repaired and checked by reading, not compiled, not run on hardware
//
#include <hip/hip_runtime.h>
#include <math.h>

// ---------------------------------------------------------------------------
// Swin transformer layer pair (window MSA + shifted window MSA) for gfx1250.
// One window (49 tokens x 192 ch) per 256-thread block; all GEMMs via
// v_wmma_f32_16x16x32_bf16; activations staged in LDS; weights streamed
// bf16/N-major from L2-resident workspace.
// ---------------------------------------------------------------------------

typedef __attribute__((ext_vector_type(16))) __bf16 v16bf;
typedef __attribute__((ext_vector_type(8)))  float  v8f;

#define WS7   7
#define SHIFT 3
#define LTOK  49
#define CDIM  192
#define HN6   6
#define DH    32
#define MPAD  64
#define HDIM  768

// ---- LDS layout (bytes), total 160512 ----
#define SM_XRES 0            // float [49][192]           37632
#define SM_XA   37632        // bf16  [64][192]           24576 (aliased by S,P)
#define SM_S    37632        // float [64][64]            16384
#define SM_P    54016        // bf16  [64][64]             8192
#define SM_Q    62208        // bf16  [64][192]           24576
#define SM_K    86784        // bf16  [64][192]           24576
#define SM_V    111360       // bf16  [64][192]           24576
#define SM_O    135936       // bf16  [64][192]           24576
#define SM_HID  62208        // bf16  [64][768]           98304 (aliases Q..O)
#define SM_TOTAL 160512

__device__ __forceinline__ v8f wmma_bf16(v16bf a, v16bf b, v8f c) {
  // v_wmma_f32_16x16x32_bf16: D = A(16x32) * B(32x16) + C
  return __builtin_amdgcn_wmma_f32_16x16x32_bf16(
      /*neg_a=*/false, a, /*neg_b=*/false, b,
      /*c_mod=*/(short)0, c, /*reuse_a=*/false, /*reuse_b=*/false);
}

// A-operand (and N-major B-operand) tile loader, 16-bit elements, 16x32 tile.
// ISA layout: lanes 0-15 hold row r=lane, K = {0..7, 16..23}; lanes 16-31 hold
// row r=lane-16, K = {8..15, 24..31}.  p0 points at element (row0, k0);
// stride = row pitch in elements.  Each half is 8 contiguous bf16 (16 bytes).
__device__ __forceinline__ v16bf ld_nmajor(const __bf16* p0, int stride, int lane) {
  const int r  = lane & 15;
  const int hf = lane >> 4;
  const __bf16* p = p0 + r * stride + 8 * hf;
  v16bf x;
#pragma unroll
  for (int e = 0; e < 8; ++e) x[e]     = p[e];
#pragma unroll
  for (int e = 0; e < 8; ++e) x[8 + e] = p[16 + e];
  return x;
}

// B-operand tile loader from a row-major (K x N) matrix, 32x16 tile.
// lanes 0-15: column n=lane, K = {0..7, 16..23} ; lanes 16-31: column n=lane-16,
// K = {8..15, 24..31}.  p0 points at element (k0, n0); stride = row pitch (elements).
__device__ __forceinline__ v16bf ldb_kmajor(const __bf16* p0, int stride, int lane) {
  const int cn = lane & 15;
  const int hf = lane >> 4;
  const __bf16* p = p0 + (8 * hf) * stride + cn;
  v16bf x;
#pragma unroll
  for (int e = 0; e < 8; ++e) x[e] = p[e * stride];
#pragma unroll
  for (int e = 0; e < 8; ++e) x[8 + e] = p[(16 + e) * stride];
  return x;
}

// ---------------------------------------------------------------------------
// prep: fp32 (K x N) weight -> bf16 (N x K) transposed
// ---------------------------------------------------------------------------
__global__ void swin_cvt_w(const float* __restrict__ src, __bf16* __restrict__ dst,
                           int K, int N) {
  for (int i = blockIdx.x * blockDim.x + threadIdx.x; i < K * N;
       i += gridDim.x * blockDim.x) {
    int k = i / N, n = i % N;
    dst[n * K + k] = (__bf16)src[i];
  }
}

// prep: combined per-head relative-position bias [+ shift mask] -> fp32 [H][L][L]
__global__ void swin_build_bias(const float* __restrict__ bias1,
                                const float* __restrict__ bias2,
                                float* __restrict__ cb1,
                                float* __restrict__ cb2) {
  const int total = HN6 * LTOK * LTOK;
  for (int i = blockIdx.x * blockDim.x + threadIdx.x; i < total;
       i += gridDim.x * blockDim.x) {
    int h = i / (LTOK * LTOK);
    int r = (i / LTOK) % LTOK;
    int c = i % LTOK;
    int qi = r / WS7, qj = r % WS7, ki = c / WS7, kj = c % WS7;
    int rel = (qi - ki + WS7 - 1) * (2 * WS7 - 1) + (qj - kj + WS7 - 1);
    cb1[i] = bias1[rel * HN6 + h];
    int rq = 2 * (qi >= WS7 - SHIFT) + (qj >= WS7 - SHIFT);
    int rk = 2 * (ki >= WS7 - SHIFT) + (kj >= WS7 - SHIFT);
    cb2[i] = bias2[rel * HN6 + h] + ((rq != rk) ? -100.0f : 0.0f);
  }
}

// ---------------------------------------------------------------------------
// fused per-window layer kernel
// ---------------------------------------------------------------------------
__global__ __launch_bounds__(256) void swin_layer_kernel(
    const float* __restrict__ Xin, float* __restrict__ Xout,
    const __bf16* __restrict__ Wqkv, const float* __restrict__ Bqkv,
    const __bf16* __restrict__ Wproj, const float* __restrict__ Bproj,
    const float* __restrict__ CB,
    const __bf16* __restrict__ W1, const float* __restrict__ B1,
    const __bf16* __restrict__ W2, const float* __restrict__ B2,
    const float* __restrict__ g1, const float* __restrict__ lb1,
    const float* __restrict__ g2, const float* __restrict__ lb2,
    int in_shift, int out_rolled) {
  extern __shared__ char smem[];
  float*  XRES = (float*)(smem + SM_XRES);
  __bf16* XA   = (__bf16*)(smem + SM_XA);
  float*  Sb   = (float*)(smem + SM_S);
  __bf16* Pb   = (__bf16*)(smem + SM_P);
  __bf16* Qb   = (__bf16*)(smem + SM_Q);
  __bf16* Kb   = (__bf16*)(smem + SM_K);
  __bf16* Vb   = (__bf16*)(smem + SM_V);
  __bf16* Ob   = (__bf16*)(smem + SM_O);
  __bf16* HID  = (__bf16*)(smem + SM_HID);

  const int tid  = threadIdx.x;
  const int lane = tid & 31;
  const int wv   = tid >> 5;
  const size_t wbase = (size_t)blockIdx.x * (LTOK * CDIM);

  // ---- phase 1: load window (optional intra-window roll on load) ----
  for (int i = tid; i < LTOK * CDIM; i += 256) {
    int m = i / CDIM, c = i % CDIM;
    int src = m;
    if (in_shift) {
      int r = m / WS7, j = m % WS7;
      src = ((r + in_shift) % WS7) * WS7 + ((j + in_shift) % WS7);
    }
    XRES[i] = Xin[wbase + (size_t)src * CDIM + c];
  }
  __syncthreads();

  // ---- phase 2: LayerNorm 1 -> XA (bf16); zero padded rows ----
  if (tid < LTOK) {
    int r = tid;
    float s = 0.f, s2 = 0.f;
    for (int c = 0; c < CDIM; ++c) {
      float x = XRES[r * CDIM + c];
      s += x; s2 += x * x;
    }
    float mn  = s * (1.0f / CDIM);
    float var = s2 * (1.0f / CDIM) - mn * mn;
    float inv = rsqrtf(var + 1e-5f);
    for (int c = 0; c < CDIM; ++c)
      XA[r * CDIM + c] = (__bf16)((XRES[r * CDIM + c] - mn) * inv * g1[c] + lb1[c]);
  } else if (tid < MPAD) {
    for (int c = 0; c < CDIM; ++c) XA[tid * CDIM + c] = (__bf16)0.f;
  }
  __syncthreads();

  // ---- phase 3: QKV GEMM (64x576 = A[64x192] * B[192x576]) ----
  {
    const float qscale = 0.17677669529663689f;  // 1/sqrt(32)
    for (int t = wv; t < 4 * 36; t += 8) {
      int mt = t / 36, nt = t % 36;
      v8f acc = {};
#pragma unroll
      for (int kt = 0; kt < 6; ++kt) {
        v16bf a = ld_nmajor(XA + (mt * 16) * CDIM + kt * 32, CDIM, lane);
        v16bf b = ld_nmajor(Wqkv + (size_t)(nt * 16) * CDIM + kt * 32, CDIM, lane);
        acc = wmma_bf16(a, b, acc);
      }
      int n  = nt * 16 + (lane & 15);
      int m0 = mt * 16 + 8 * (lane >> 4);
      float bias = Bqkv[n];
      __bf16* dst; int nc; float sc;
      if (nt < 12)      { dst = Qb; nc = n;            sc = qscale; }
      else if (nt < 24) { dst = Kb; nc = n - CDIM;     sc = 1.0f;   }
      else              { dst = Vb; nc = n - 2 * CDIM; sc = 1.0f;   }
#pragma unroll
      for (int v = 0; v < 8; ++v)
        dst[(m0 + v) * CDIM + nc] = (__bf16)((acc[v] + bias) * sc);
    }
  }
  __syncthreads();

  // zero probability pad rows once (P aliases XA region; XA reads are done)
  for (int i = tid; i < MPAD * MPAD; i += 256) Pb[i] = (__bf16)0.f;

  // ---- phase 4: attention, one head at a time ----
  for (int h = 0; h < HN6; ++h) {
    // S = (q*scale) @ k^T : 4x4 tiles, K = 32 (one wmma each)
    for (int t = wv; t < 16; t += 8) {
      int mt = t >> 2, nt = t & 3;
      v8f acc = {};
      v16bf a = ld_nmajor(Qb + (mt * 16) * CDIM + h * DH, CDIM, lane);
      v16bf b = ld_nmajor(Kb + (nt * 16) * CDIM + h * DH, CDIM, lane);  // == K^T tile
      acc = wmma_bf16(a, b, acc);
      int n  = nt * 16 + (lane & 15);
      int m0 = mt * 16 + 8 * (lane >> 4);
#pragma unroll
      for (int v = 0; v < 8; ++v) Sb[(m0 + v) * MPAD + n] = acc[v];
    }
    __syncthreads();

    // softmax over valid 49 columns, with combined bias(+mask) from L2
    if (tid < LTOK) {
      int r = tid;
      const float* cb = CB + ((size_t)h * LTOK + r) * LTOK;
      float mx = -1e30f;
      for (int c = 0; c < LTOK; ++c) mx = fmaxf(mx, Sb[r * MPAD + c] + cb[c]);
      float sum = 0.f;
      for (int c = 0; c < LTOK; ++c) sum += __expf(Sb[r * MPAD + c] + cb[c] - mx);
      float inv = 1.0f / sum;
      for (int c = 0; c < LTOK; ++c)
        Pb[r * MPAD + c] = (__bf16)(__expf(Sb[r * MPAD + c] + cb[c] - mx) * inv);
      for (int c = LTOK; c < MPAD; ++c) Pb[r * MPAD + c] = (__bf16)0.f;
    }
    __syncthreads();

    // O[:, h*32 .. h*32+31] = P(64x64) @ V_h(64x32) : 4x2 tiles, K = 2
    for (int t = wv; t < 8; t += 8) {
      int mt = t >> 1, nt = t & 1;
      v8f acc = {};
#pragma unroll
      for (int kt = 0; kt < 2; ++kt) {
        v16bf a = ld_nmajor(Pb + (mt * 16) * MPAD + kt * 32, MPAD, lane);
        v16bf b = ldb_kmajor(Vb + (kt * 32) * CDIM + h * DH + nt * 16, CDIM, lane);
        acc = wmma_bf16(a, b, acc);
      }
      int n  = h * DH + nt * 16 + (lane & 15);
      int m0 = mt * 16 + 8 * (lane >> 4);
#pragma unroll
      for (int v = 0; v < 8; ++v) Ob[(m0 + v) * CDIM + n] = (__bf16)acc[v];
    }
    __syncthreads();
  }

  // ---- phase 5: projection + residual into XRES ----
  for (int t = wv; t < 4 * 12; t += 8) {
    int mt = t / 12, nt = t % 12;
    v8f acc = {};
#pragma unroll
    for (int kt = 0; kt < 6; ++kt) {
      v16bf a = ld_nmajor(Ob + (mt * 16) * CDIM + kt * 32, CDIM, lane);
      v16bf b = ld_nmajor(Wproj + (size_t)(nt * 16) * CDIM + kt * 32, CDIM, lane);
      acc = wmma_bf16(a, b, acc);
    }
    int n  = nt * 16 + (lane & 15);
    int m0 = mt * 16 + 8 * (lane >> 4);
    float bias = Bproj[n];
#pragma unroll
    for (int v = 0; v < 8; ++v) {
      int m = m0 + v;
      if (m < LTOK) XRES[m * CDIM + n] += acc[v] + bias;
    }
  }
  __syncthreads();

  // ---- phase 6: LayerNorm 2 (X = ln2(X + attn)) -> XRES fp32 + XA bf16 ----
  if (tid < LTOK) {
    int r = tid;
    float s = 0.f, s2 = 0.f;
    for (int c = 0; c < CDIM; ++c) {
      float x = XRES[r * CDIM + c];
      s += x; s2 += x * x;
    }
    float mn  = s * (1.0f / CDIM);
    float var = s2 * (1.0f / CDIM) - mn * mn;
    float inv = rsqrtf(var + 1e-5f);
    for (int c = 0; c < CDIM; ++c) {
      float y = (XRES[r * CDIM + c] - mn) * inv * g2[c] + lb2[c];
      XRES[r * CDIM + c] = y;
      XA[r * CDIM + c]   = (__bf16)y;
    }
  } else if (tid < MPAD) {
    for (int c = 0; c < CDIM; ++c) XA[tid * CDIM + c] = (__bf16)0.f;  // re-zero (S/P aliased)
  }
  __syncthreads();

  // ---- phase 7: MLP up-projection + exact GELU -> HID bf16 [64][768] ----
  for (int t = wv; t < 4 * 48; t += 8) {
    int mt = t / 48, nt = t % 48;
    v8f acc = {};
#pragma unroll
    for (int kt = 0; kt < 6; ++kt) {
      v16bf a = ld_nmajor(XA + (mt * 16) * CDIM + kt * 32, CDIM, lane);
      v16bf b = ld_nmajor(W1 + (size_t)(nt * 16) * CDIM + kt * 32, CDIM, lane);
      acc = wmma_bf16(a, b, acc);
    }
    int n  = nt * 16 + (lane & 15);
    int m0 = mt * 16 + 8 * (lane >> 4);
    float bias = B1[n];
#pragma unroll
    for (int v = 0; v < 8; ++v) {
      float x = acc[v] + bias;
      HID[(m0 + v) * HDIM + n] = (__bf16)(0.5f * x * (1.0f + erff(x * 0.70710678118654752f)));
    }
  }
  __syncthreads();

  // ---- phase 8: MLP down-projection + residual, store (optional reverse roll) ----
  for (int t = wv; t < 4 * 12; t += 8) {
    int mt = t / 12, nt = t % 12;
    v8f acc = {};
    for (int kt = 0; kt < 24; ++kt) {
      v16bf a = ld_nmajor(HID + (mt * 16) * HDIM + kt * 32, HDIM, lane);
      v16bf b = ld_nmajor(W2 + (size_t)(nt * 16) * HDIM + kt * 32, HDIM, lane);
      acc = wmma_bf16(a, b, acc);
    }
    int n  = nt * 16 + (lane & 15);
    int m0 = mt * 16 + 8 * (lane >> 4);
    float bias = B2[n];
#pragma unroll
    for (int v = 0; v < 8; ++v) {
      int m = m0 + v;
      if (m < LTOK) {
        float y = XRES[m * CDIM + n] + acc[v] + bias;
        int i = m / WS7, j = m % WS7, ch = n;
        if (out_rolled) {  // reverse roll on (W, C) axes, replicated faithfully
          j  = (j + SHIFT) % WS7;
          ch = (n + SHIFT) % CDIM;
        }
        Xout[wbase + (size_t)(i * WS7 + j) * CDIM + ch] = y;
      }
    }
  }
}

// ---------------------------------------------------------------------------
// host side
// ---------------------------------------------------------------------------
extern "C" void kernel_launch(void* const* d_in, const int* in_sizes, int n_in,
                              void* d_out, int out_size, void* d_ws, size_t ws_size,
                              hipStream_t stream) {
  const float* X        = (const float*)d_in[0];
  const float* ln1_g    = (const float*)d_in[1];
  const float* ln1_b    = (const float*)d_in[2];
  const float* ln2_g    = (const float*)d_in[3];
  const float* ln2_b    = (const float*)d_in[4];
  const float* wqkv1    = (const float*)d_in[5];
  const float* bqkv1    = (const float*)d_in[6];
  const float* wproj1   = (const float*)d_in[7];
  const float* bproj1   = (const float*)d_in[8];
  const float* wbias1   = (const float*)d_in[9];
  const float* wqkv2    = (const float*)d_in[10];
  const float* bqkv2    = (const float*)d_in[11];
  const float* wproj2   = (const float*)d_in[12];
  const float* bproj2   = (const float*)d_in[13];
  const float* wbias2   = (const float*)d_in[14];
  const float* w1       = (const float*)d_in[15];
  const float* b1       = (const float*)d_in[16];
  const float* w2       = (const float*)d_in[17];
  const float* b2       = (const float*)d_in[18];
  float* out = (float*)d_out;

  // workspace carve-up (bf16 transposed weights + combined biases)
  char* ws = (char*)d_ws;
  __bf16* Wqkv1t  = (__bf16*)(ws + 0);        // 576*192*2 = 221184
  __bf16* Wproj1t = (__bf16*)(ws + 221184);   // 192*192*2 =  73728
  __bf16* Wqkv2t  = (__bf16*)(ws + 294912);   //             221184
  __bf16* Wproj2t = (__bf16*)(ws + 516096);   //              73728
  __bf16* W1t     = (__bf16*)(ws + 589824);   // 768*192*2 = 294912
  __bf16* W2t     = (__bf16*)(ws + 884736);   // 192*768*2 = 294912
  float*  CB1     = (float*)(ws + 1179648);   // 6*49*49*4 =  57624
  float*  CB2     = (float*)(ws + 1237272);   //              57624

  // weight prep (all on stream; graph-capture safe)
  swin_cvt_w<<<dim3(128), dim3(256), 0, stream>>>(wqkv1,  Wqkv1t,  CDIM, 3 * CDIM);
  swin_cvt_w<<<dim3(64),  dim3(256), 0, stream>>>(wproj1, Wproj1t, CDIM, CDIM);
  swin_cvt_w<<<dim3(128), dim3(256), 0, stream>>>(wqkv2,  Wqkv2t,  CDIM, 3 * CDIM);
  swin_cvt_w<<<dim3(64),  dim3(256), 0, stream>>>(wproj2, Wproj2t, CDIM, CDIM);
  swin_cvt_w<<<dim3(128), dim3(256), 0, stream>>>(w1,     W1t,     CDIM, HDIM);
  swin_cvt_w<<<dim3(128), dim3(256), 0, stream>>>(w2,     W2t,     HDIM, CDIM);
  swin_build_bias<<<dim3(64), dim3(256), 0, stream>>>(wbias1, wbias2, CB1, CB2);

  const int nwin = in_sizes[0] / (LTOK * CDIM);  // 8 * 1024 = 8192 windows
  (void)hipFuncSetAttribute((const void*)swin_layer_kernel,
                            hipFuncAttributeMaxDynamicSharedMemorySize, SM_TOTAL);

  // block 1: plain window MSA (no input shift, no output roll) -> d_out
  swin_layer_kernel<<<dim3(nwin), dim3(256), SM_TOTAL, stream>>>(
      X, out, Wqkv1t, bqkv1, Wproj1t, bproj1, CB1, W1t, b1, W2t, b2,
      ln1_g, ln1_b, ln2_g, ln2_b, /*in_shift=*/0, /*out_rolled=*/0);

  // block 2: shifted window MSA (roll on load, reverse roll on store), in-place
  swin_layer_kernel<<<dim3(nwin), dim3(256), SM_TOTAL, stream>>>(
      out, out, Wqkv2t, bqkv2, Wproj2t, bproj2, CB2, W1t, b1, W2t, b2,
      ln1_g, ln1_b, ln2_g, ln2_b, /*in_shift=*/SHIFT, /*out_rolled=*/1);

  (void)n_in; (void)out_size; (void)ws_size;
}
